// VideoCaptionDecoder_81406810129221
// MI455X (gfx1250) — matrix operand order repaired and checked by reading, not compiled, not run on hardware
//
#include <hip/hip_runtime.h>
#include <hip/hip_bf16.h>
#include <math.h>

// Problem dims (fixed by the reference)
#define BATCH 64
#define TLEN  20
#define SLEN  40
#define DW    512
#define DE    512
#define HD    512
#define AD    512
#define VV    20000

typedef __attribute__((ext_vector_type(16))) __bf16 v16bf;
typedef __attribute__((ext_vector_type(8)))  __bf16 bf16x8;
typedef __attribute__((ext_vector_type(8)))  float  v8f;

union BFrag { v16bf v; bf16x8 h[2]; };

// ---------------------------------------------------------------------------
// bf16 WMMA GEMM:  C[M,N] (f32) = A[M,K](bf16, row-major, stride lda)
//                               * B[N,K](bf16, row-major => implicit B^T)
//                               + bias[N] (optional)
// Each wave computes a 64x16 slab (4 M-subtiles share one B fragment),
// K-loop in steps of 32 via v_wmma_f32_16x16x32_bf16. M % 64 == 0,
// N % 16 == 0, K % 32 == 0 (all shapes here satisfy this).
// ---------------------------------------------------------------------------
__global__ void k_gemm_bf16(const __bf16* __restrict__ A, int lda,
                            const __bf16* __restrict__ B,
                            const float* __restrict__ bias,
                            float* __restrict__ C, long long ldc,
                            int M, int N, int K)
{
    const int wave = blockIdx.x * (blockDim.x >> 5) + (threadIdx.x >> 5);
    const int lane = threadIdx.x & 31;
    const int ntiles  = N >> 4;
    const int mblocks = M >> 6;
    if (wave >= mblocks * ntiles) return;   // wave-uniform: EXEC stays all-1s

    const int nt  = wave % ntiles;
    const int mb  = wave / ntiles;
    const int col  = nt * 16 + (lane & 15);
    const int half = lane >> 4;

    // B fragment: elements j=0..7 <-> K = kb + half*8 + j; j=8..15 <-> K = kb + 16 + half*8 + (j-8)
    const __bf16* Brow = B + (long long)col * K + half * 8;
    // A fragment: elements j=0..7 <-> K = kb + half*8 + j
    //             elements j=8..15 <-> K = kb + 16 + half*8 + (j-8)
    const long long arow = (long long)(mb * 64 + (lane & 15)) * lda + half * 8;
    const __bf16* A0 = A + arow;
    const __bf16* A1 = A0 + (long long)16 * lda;
    const __bf16* A2 = A0 + (long long)32 * lda;
    const __bf16* A3 = A0 + (long long)48 * lda;

    v8f acc0 = {}, acc1 = {}, acc2 = {}, acc3 = {};

    for (int kb = 0; kb < K; kb += 32) {
        BFrag bf, a0, a1, a2, a3;
        bf.h[0] = *(const bf16x8*)(Brow + kb);
        bf.h[1] = *(const bf16x8*)(Brow + kb + 16);
        a0.h[0] = *(const bf16x8*)(A0 + kb);
        a0.h[1] = *(const bf16x8*)(A0 + kb + 16);
        a1.h[0] = *(const bf16x8*)(A1 + kb);
        a1.h[1] = *(const bf16x8*)(A1 + kb + 16);
        a2.h[0] = *(const bf16x8*)(A2 + kb);
        a2.h[1] = *(const bf16x8*)(A2 + kb + 16);
        a3.h[0] = *(const bf16x8*)(A3 + kb);
        a3.h[1] = *(const bf16x8*)(A3 + kb + 16);
        acc0 = __builtin_amdgcn_wmma_f32_16x16x32_bf16(false, a0.v, false, bf.v,
                                                       (short)0, acc0, false, false);
        acc1 = __builtin_amdgcn_wmma_f32_16x16x32_bf16(false, a1.v, false, bf.v,
                                                       (short)0, acc1, false, false);
        acc2 = __builtin_amdgcn_wmma_f32_16x16x32_bf16(false, a2.v, false, bf.v,
                                                       (short)0, acc2, false, false);
        acc3 = __builtin_amdgcn_wmma_f32_16x16x32_bf16(false, a3.v, false, bf.v,
                                                       (short)0, acc3, false, false);
    }

    const float bv = bias ? bias[col] : 0.0f;
    const int rbase = mb * 64 + half * 8;
    #pragma unroll
    for (int r = 0; r < 8; ++r) {
        C[(long long)(rbase +  0 + r) * ldc + col] = acc0[r] + bv;
        C[(long long)(rbase + 16 + r) * ldc + col] = acc1[r] + bv;
        C[(long long)(rbase + 32 + r) * ldc + col] = acc2[r] + bv;
        C[(long long)(rbase + 48 + r) * ldc + col] = acc3[r] + bv;
    }
}

// ---------------------------------------------------------------------------
// Weight / data prep kernels
// ---------------------------------------------------------------------------
__global__ void k_cat2_bf16(__bf16* __restrict__ dst, const float* __restrict__ W1,
                            const float* __restrict__ W2, int N, int K1, int K2)
{
    const int K = K1 + K2;
    const long long total = (long long)N * K;
    for (long long i = blockIdx.x * (long long)blockDim.x + threadIdx.x;
         i < total; i += (long long)gridDim.x * blockDim.x) {
        int n = (int)(i / K), k = (int)(i % K);
        float v = (k < K1) ? W1[(long long)n * K1 + k]
                           : W2[(long long)n * K2 + (k - K1)];
        dst[i] = (__bf16)v;
    }
}

__global__ void k_transpose_bf16(__bf16* __restrict__ dst /*[N,K]*/,
                                 const float* __restrict__ src /*[K,N]*/,
                                 int N, int K)
{
    const long long total = (long long)N * K;
    for (long long i = blockIdx.x * (long long)blockDim.x + threadIdx.x;
         i < total; i += (long long)gridDim.x * blockDim.x) {
        int n = (int)(i / K), k = (int)(i % K);
        dst[i] = (__bf16)src[(long long)k * N + n];
    }
}

__global__ void k_f32_to_bf16(__bf16* __restrict__ dst, const float* __restrict__ src,
                              long long n)
{
    for (long long i = blockIdx.x * (long long)blockDim.x + threadIdx.x;
         i < n; i += (long long)gridDim.x * blockDim.x)
        dst[i] = (__bf16)src[i];
}

__global__ void k_add_f32(float* __restrict__ dst, const float* __restrict__ a,
                          const float* __restrict__ b, int n)
{
    int i = blockIdx.x * blockDim.x + threadIdx.x;
    if (i < n) dst[i] = a[i] + b[i];
}

__global__ void k_zero_f32(float* __restrict__ p, int n)
{
    int i = blockIdx.x * blockDim.x + threadIdx.x;
    if (i < n) p[i] = 0.0f;
}

__global__ void k_zero_bf16_cols(__bf16* __restrict__ base, int stride,
                                 int col0, int ncols, int rows)
{
    int total = rows * ncols;
    int i = blockIdx.x * blockDim.x + threadIdx.x;
    if (i >= total) return;
    int r = i / ncols, c = i % ncols;
    base[(long long)r * stride + col0 + c] = (__bf16)0.0f;
}

// x_t (f32 [B,T,DW] slice) -> abuf_b[:, 0:DW] bf16
__global__ void k_copy_x(__bf16* __restrict__ abuf_b, const float* __restrict__ iseq, int t)
{
    int i = blockIdx.x * blockDim.x + threadIdx.x;
    if (i >= BATCH * DW) return;
    int b = i / DW, d = i % DW;
    abuf_b[(long long)b * 1024 + d] =
        (__bf16)iseq[((long long)b * TLEN + t) * DW + d];
}

// ---------------------------------------------------------------------------
// LSTM elementwise: gates f32 [B,4H] (i,f,g,o), c f32 [B,H] in/out,
// h written as bf16 into up to 3 activation buffers (free concat).
// ---------------------------------------------------------------------------
__global__ void k_lstm_cell(const float* __restrict__ gates, float* __restrict__ c,
                            __bf16* h0, int s0, int o0,
                            __bf16* h1, int s1, int o1,
                            __bf16* h2, int s2, int o2)
{
    int idx = blockIdx.x * blockDim.x + threadIdx.x;
    if (idx >= BATCH * HD) return;
    int b = idx / HD, j = idx % HD;
    const float* g = gates + (long long)b * 4 * HD;
    float gi = g[j], gf = g[HD + j], gg = g[2 * HD + j], go = g[3 * HD + j];
    float si = 1.0f / (1.0f + expf(-gi));
    float sf = 1.0f / (1.0f + expf(-gf));
    float so = 1.0f / (1.0f + expf(-go));
    float cn = sf * c[idx] + si * tanhf(gg);
    float hn = so * tanhf(cn);
    c[idx] = cn;
    __bf16 hb = (__bf16)hn;
    h0[(long long)b * s0 + o0 + j] = hb;
    if (h1) h1[(long long)b * s1 + o1 + j] = hb;
    if (h2) h2[(long long)b * s2 + o2 + j] = hb;
}

// ---------------------------------------------------------------------------
// Attention: one block per batch row.
//   e[s] = sum_a tanh(es_proj[b,s,a] + q[b,a]) * va[a]
//   attn = softmax(mask(e)); ctx[d] = sum_s attn[s]*es[b,s,d]  -> bf16 slot
// ---------------------------------------------------------------------------
__global__ void k_attention(const float* __restrict__ es_proj, const float* __restrict__ q,
                            const float* __restrict__ va, const float* __restrict__ es,
                            const int* __restrict__ lens,
                            __bf16* __restrict__ ctx_out, int ctx_stride)
{
    __shared__ float e_l[SLEN];
    __shared__ float a_l[SLEN];
    const int b = blockIdx.x;
    const int tid = threadIdx.x, wave = tid >> 5, lane = tid & 31;

    const float* qp = q + (long long)b * AD;
    for (int s = wave; s < SLEN; s += 8) {
        const float* ep = es_proj + ((long long)b * SLEN + s) * AD;
        float p = 0.0f;
        for (int a = lane; a < AD; a += 32)
            p += tanhf(ep[a] + qp[a]) * va[a];
        for (int off = 16; off > 0; off >>= 1)
            p += __shfl_xor(p, off, 32);
        if (lane == 0) e_l[s] = p;
    }
    __syncthreads();

    const int len = lens[b];
    float mx = -1e30f;
    for (int s = 0; s < SLEN; ++s) {
        float v = (s < len) ? e_l[s] : -1e9f;
        mx = fmaxf(mx, v);
    }
    float sum = 0.0f;
    for (int s = 0; s < SLEN; ++s) {
        float v = (s < len) ? e_l[s] : -1e9f;
        sum += expf(v - mx);
    }
    if (tid < SLEN)
        a_l[tid] = (tid < len) ? expf(e_l[tid] - mx) / sum : 0.0f;
    __syncthreads();

    for (int d = tid; d < DE; d += blockDim.x) {
        float cv = 0.0f;
        for (int s = 0; s < SLEN; ++s)
            cv += a_l[s] * es[((long long)b * SLEN + s) * DE + d];
        ctx_out[(long long)b * ctx_stride + d] = (__bf16)cv;
    }
}

// In-place log-softmax over rows of V
__global__ void k_log_softmax(float* __restrict__ out, int V)
{
    __shared__ float red[256];
    float* p = out + (long long)blockIdx.x * V;
    const int tid = threadIdx.x;

    float mx = -1e30f;
    for (int i = tid; i < V; i += blockDim.x) mx = fmaxf(mx, p[i]);
    red[tid] = mx; __syncthreads();
    for (int s = 128; s > 0; s >>= 1) {
        if (tid < s) red[tid] = fmaxf(red[tid], red[tid + s]);
        __syncthreads();
    }
    mx = red[0]; __syncthreads();

    float sum = 0.0f;
    for (int i = tid; i < V; i += blockDim.x) sum += expf(p[i] - mx);
    red[tid] = sum; __syncthreads();
    for (int s = 128; s > 0; s >>= 1) {
        if (tid < s) red[tid] += red[tid + s];
        __syncthreads();
    }
    float lse = mx + logf(red[0]);
    __syncthreads();

    for (int i = tid; i < V; i += blockDim.x) p[i] -= lse;
}

// ---------------------------------------------------------------------------
// Host side
// ---------------------------------------------------------------------------
static inline size_t align_up(size_t x) { return (x + 255) & ~(size_t)255; }

static void launch_gemm(const __bf16* A, int lda, const __bf16* B, const float* bias,
                        float* C, long long ldc, int M, int N, int K, hipStream_t s)
{
    int waves  = (M / 64) * (N / 16);
    int blocks = (waves + 7) / 8;           // 8 waves (256 threads) per block
    k_gemm_bf16<<<blocks, 256, 0, s>>>(A, lda, B, bias, C, ldc, M, N, K);
}

static inline int gsz(long long n) {
    long long b = (n + 255) / 256;
    return (int)(b > 8192 ? 8192 : b);
}

extern "C" void kernel_launch(void* const* d_in, const int* in_sizes, int n_in,
                              void* d_out, int out_size, void* d_ws, size_t ws_size,
                              hipStream_t stream)
{
    (void)in_sizes; (void)n_in; (void)out_size; (void)ws_size;
    const float* iseq  = (const float*)d_in[0];
    const float* eseq  = (const float*)d_in[1];
    const int*   lens  = (const int*)  d_in[2];
    const float* Wih_b = (const float*)d_in[3];
    const float* Whh_b = (const float*)d_in[4];
    const float* bih_b = (const float*)d_in[5];
    const float* bhh_b = (const float*)d_in[6];
    const float* Wih_t = (const float*)d_in[7];
    const float* Whh_t = (const float*)d_in[8];
    const float* bih_t = (const float*)d_in[9];
    const float* bhh_t = (const float*)d_in[10];
    const float* Wa    = (const float*)d_in[11];
    const float* Ua    = (const float*)d_in[12];
    const float* va    = (const float*)d_in[13];
    const float* Wout  = (const float*)d_in[14];
    const float* bout  = (const float*)d_in[15];
    float* out = (float*)d_out;

    // ---- workspace carve (~45 MB; fits comfortably in L2-resident working set)
    char* ws = (char*)d_ws;
    size_t off = 0;
    auto carve = [&](size_t bytes) { char* p = ws + off; off = align_up(off + bytes); return p; };
    __bf16* Wb_cat  = (__bf16*)carve((size_t)2048 * 1024 * 2);  // [Wih_b|Whh_b]
    __bf16* Wt_cat  = (__bf16*)carve((size_t)2048 * 1536 * 2);  // [Wih_t|Whh_t]
    __bf16* UaT     = (__bf16*)carve((size_t)512 * 1024 * 2);   // Ua^T  [A,2H]
    __bf16* WaT     = (__bf16*)carve((size_t)512 * 512 * 2);    // Wa^T  [A,DE]
    __bf16* Wout_bf = (__bf16*)carve((size_t)VV * HD * 2);
    __bf16* es_bf   = (__bf16*)carve((size_t)BATCH * SLEN * DE * 2);
    float*  es_proj = (float*) carve((size_t)BATCH * SLEN * AD * 4);
    float*  bias_b  = (float*) carve((size_t)2048 * 4);
    float*  bias_t  = (float*) carve((size_t)2048 * 4);
    __bf16* abuf_b  = (__bf16*)carve((size_t)BATCH * 1024 * 2); // [x_t | h_b]
    __bf16* abuf_q  = (__bf16*)carve((size_t)BATCH * 1024 * 2); // [h_b | h_t]
    __bf16* abuf_t  = (__bf16*)carve((size_t)BATCH * 1536 * 2); // [ctx | h_b | h_t]
    float*  gates   = (float*) carve((size_t)BATCH * 2048 * 4);
    float*  qbuf    = (float*) carve((size_t)BATCH * AD * 4);
    float*  c_b     = (float*) carve((size_t)BATCH * HD * 4);
    float*  c_t     = (float*) carve((size_t)BATCH * HD * 4);

    // ---- one-time per call: weight prep (deterministic, no caching)
    k_cat2_bf16<<<gsz((long long)2048 * 1024), 256, 0, stream>>>(Wb_cat, Wih_b, Whh_b, 2048, 512, 512);
    k_cat2_bf16<<<gsz((long long)2048 * 1536), 256, 0, stream>>>(Wt_cat, Wih_t, Whh_t, 2048, 1024, 512);
    k_transpose_bf16<<<gsz((long long)512 * 1024), 256, 0, stream>>>(UaT, Ua, 512, 1024);
    k_transpose_bf16<<<gsz((long long)512 * 512),  256, 0, stream>>>(WaT, Wa, 512, 512);
    k_f32_to_bf16<<<gsz((long long)VV * HD), 256, 0, stream>>>(Wout_bf, Wout, (long long)VV * HD);
    k_f32_to_bf16<<<gsz((long long)BATCH * SLEN * DE), 256, 0, stream>>>(es_bf, eseq, (long long)BATCH * SLEN * DE);
    k_add_f32<<<8, 256, 0, stream>>>(bias_b, bih_b, bhh_b, 2048);
    k_add_f32<<<8, 256, 0, stream>>>(bias_t, bih_t, bhh_t, 2048);

    // ---- zero recurrent state (ws is poisoned by the harness)
    k_zero_f32<<<gsz(BATCH * HD), 256, 0, stream>>>(c_b, BATCH * HD);
    k_zero_f32<<<gsz(BATCH * HD), 256, 0, stream>>>(c_t, BATCH * HD);
    k_zero_bf16_cols<<<gsz(BATCH * 512),  256, 0, stream>>>(abuf_b, 1024, 512, 512,  BATCH);
    k_zero_bf16_cols<<<gsz(BATCH * 1024), 256, 0, stream>>>(abuf_q, 1024, 0,   1024, BATCH);
    k_zero_bf16_cols<<<gsz(BATCH * 1024), 256, 0, stream>>>(abuf_t, 1536, 512, 1024, BATCH);

    // ---- es_proj = es @ Wa   (M = B*S = 2560, N = A, K = DE)
    launch_gemm(es_bf, DE, WaT, nullptr, es_proj, AD, BATCH * SLEN, AD, DE, stream);

    // ---- time loop
    for (int t = 0; t < TLEN; ++t) {
        k_copy_x<<<gsz(BATCH * DW), 256, 0, stream>>>(abuf_b, iseq, t);

        // bottom gates: [x|h_b] @ [Wih_b|Whh_b]^T + b
        launch_gemm(abuf_b, 1024, Wb_cat, bias_b, gates, 2048, BATCH, 2048, 1024, stream);
        k_lstm_cell<<<gsz(BATCH * HD), 256, 0, stream>>>(gates, c_b,
            abuf_b, 1024, 512,    // h_b for next bottom step
            abuf_q, 1024, 0,      // h_b for attention query
            abuf_t, 1536, 512);   // h_b for top LSTM input

        // q = [h_b|h_t] @ Ua
        launch_gemm(abuf_q, 1024, UaT, nullptr, qbuf, AD, BATCH, AD, 1024, stream);

        // attention -> ctx (bf16 into abuf_t cols 0..511)
        k_attention<<<BATCH, 256, 0, stream>>>(es_proj, qbuf, va, eseq, lens, abuf_t, 1536);

        // top gates: [ctx|h_b|h_t] @ [Wih_t|Whh_t]^T + b
        launch_gemm(abuf_t, 1536, Wt_cat, bias_t, gates, 2048, BATCH, 2048, 1536, stream);
        k_lstm_cell<<<gsz(BATCH * HD), 256, 0, stream>>>(gates, c_t,
            abuf_q, 1024, 512,    // h_t for attention query
            abuf_t, 1536, 1024,   // h_t for next top step
            nullptr, 0, 0);

        // logits: h_t @ Wout^T + bout, written straight into d_out[:, t, :]
        launch_gemm(abuf_q + 512, 1024, Wout_bf, bout,
                    out + (long long)t * VV, (long long)TLEN * VV,
                    BATCH, VV, HD, stream);
    }

    // ---- final log-softmax over V, in place
    k_log_softmax<<<BATCH * TLEN, 256, 0, stream>>>(out, VV);
}
